// ADAM_17575006175639
// MI455X (gfx1250) — compile-verified, matrix operand order checked by reading, not run on hardware
//
#include <hip/hip_runtime.h>
#include <hip/hip_bf16.h>
#include <math.h>

// ---------------- problem constants ----------------
constexpr int Bn   = 16;
constexpr int Cc   = 512;
constexpr int Hh   = 64;
constexpr int Ww   = 64;
constexpr int CRr  = 28;
constexpr int K7   = 7;
constexpr int PADc = 3;
constexpr int KK   = K7 * K7;        // 49
constexpr int OFFC = 2 * KK;         // 98
constexpr int HW   = Hh * Ww;        // 4096
constexpr int KD   = CRr * KK;       // 1372 (im2col K for offset conv)

// ---------------- padded (WMMA-friendly) sizes ----------------
constexpr int CRP  = 32;             // 28 -> 32 channels
constexpr int OFFP = 112;            // 98 -> 112 (7 N-tiles)
constexpr int KDP  = 1376;           // 1372 -> 43 K-steps of 32

// ---------------- workspace layout (float units) ----------------
constexpr size_t WS_XR    = 0;                                    // [Bn][CRP][HW] f32
constexpr size_t WS_PAVG  = WS_XR    + (size_t)Bn * CRP * HW;     // [Bn][CRP]
constexpr size_t WS_PMAX  = WS_PAVG  + (size_t)Bn * CRP;          // [Bn]
constexpr size_t WS_SE    = WS_PMAX  + (size_t)Bn;                // [Bn][CRP]
constexpr size_t WS_OFF   = WS_SE    + (size_t)Bn * CRP;          // [Bn][OFFP][HW] f32
constexpr size_t WS_SA    = WS_OFF   + (size_t)Bn * OFFP * HW;    // [Bn][CRP][HW] f32
constexpr size_t WS_BNSUM = WS_SA    + (size_t)Bn * CRP * HW;     // [CRP]
constexpr size_t WS_BNSQ  = WS_BNSUM + (size_t)CRP;
constexpr size_t WS_BNSC  = WS_BNSQ  + (size_t)CRP;
constexpr size_t WS_BNSH  = WS_BNSC  + (size_t)CRP;
// bf16 padded weight copies (element counts halved into float units)
constexpr size_t WS_WOFFB = WS_BNSH  + (size_t)CRP;               // [OFFP][KDP]  bf16
constexpr size_t WS_WDCNB = WS_WOFFB + (size_t)OFFP * KDP / 2;    // [KK][CRP][CRP] bf16
constexpr size_t WS_WRESB = WS_WDCNB + (size_t)KK * CRP * CRP / 2;// [Cc][CRP]    bf16
constexpr size_t WS_WREDB = WS_WRESB + (size_t)Cc * CRP / 2;      // [CRP][Cc]    bf16
constexpr size_t WS_END   = WS_WREDB + (size_t)CRP * Cc / 2;

// ---------------- WMMA plumbing (CDNA5, wave32) ----------------
typedef __attribute__((ext_vector_type(16))) __bf16 v16bf;
typedef __attribute__((ext_vector_type(8)))  __bf16 v8bf;
typedef __attribute__((ext_vector_type(8)))  float  v8f;
typedef int v4i_gcc __attribute__((vector_size(16)));   // matches async-LDS builtin proto

__device__ __forceinline__ int lane_id() { return threadIdx.x & 31; }

// K index inside a 32-wide K step for fragment element e, given lane-half.
// 16-bit A/B layout: lanes 0-15 hold K {0..7,16..23}, lanes 16-31 hold K {8..15,24..31}.
__device__ __forceinline__ int kmap(int e, int khalf) {
  return (e < 8 ? e : 8 + e) + khalf * 8;
}

// Per-lane B fragment = two contiguous, 16B-aligned 8x bf16 runs:
// elements 0..7  <- base[khf*8      .. khf*8+7]
// elements 8..15 <- base[16 + khf*8 .. 16+khf*8+7]
template <typename PTR>
__device__ __forceinline__ v16bf load_bfrag(PTR base, int khf) {
  v8bf lo = *(const v8bf*)(base + khf * 8);
  v8bf hi = *(const v8bf*)(base + 16 + khf * 8);
  return __builtin_shufflevector(lo, hi, 0, 1, 2, 3, 4, 5, 6, 7,
                                 8, 9, 10, 11, 12, 13, 14, 15);
}

__device__ __forceinline__ v8f wmma_bf16(v16bf a, v16bf b, v8f c) {
  // D(MxN,f32) = A(16x32,bf16) * B(32x16,bf16) + C
  return __builtin_amdgcn_wmma_f32_16x16x32_bf16(
      /*neg_a=*/false, a, /*neg_b=*/false, b,
      /*c_mod=*/(short)0, c, /*reuse_a=*/false, /*reuse_b=*/false);
}

#if defined(__gfx1250__) && __has_builtin(__builtin_amdgcn_global_load_async_to_lds_b128)
#define HAVE_ASYNC_LDS 1
#else
#define HAVE_ASYNC_LDS 0
#endif

// ---------------- kernel 0: zero accumulators + SE pad lanes ---------------
__global__ void zero_acc_kernel(float* __restrict__ ws) {
  int i = blockIdx.x * blockDim.x + threadIdx.x;
  if (i < Bn * CRP) ws[WS_PAVG + i] = 0.f;
  if (i < Bn * CRP) ws[WS_SE + i] = 0.f;
  if (i < Bn)       ws[WS_PMAX + i] = 0.f;
  if (i < CRP) { ws[WS_BNSUM + i] = 0.f; ws[WS_BNSQ + i] = 0.f; }
}

// ---------------- kernel 0b: zero-padded bf16 weight copies ----------------
__global__ void pad_weights_kernel(const float* __restrict__ w_off,
                                   const float* __restrict__ w_dcn,
                                   const float* __restrict__ w_res,
                                   const float* __restrict__ w_red,
                                   float* __restrict__ ws) {
  constexpr int N_OFF = OFFP * KDP;          // 154,112
  constexpr int N_DCN = KK * CRP * CRP;      // 50,176
  constexpr int N_RES = Cc * CRP;            // 16,384
  constexpr int N_RED = CRP * Cc;            // 16,384
  __bf16* wof = (__bf16*)(ws + WS_WOFFB);
  __bf16* wdc = (__bf16*)(ws + WS_WDCNB);
  __bf16* wrs = (__bf16*)(ws + WS_WRESB);
  __bf16* wrd = (__bf16*)(ws + WS_WREDB);
  int i = blockIdx.x * blockDim.x + threadIdx.x;
  if (i < N_OFF) {
    int n = i / KDP, k = i % KDP;
    wof[i] = (__bf16)((n < OFFC && k < KD) ? w_off[(size_t)n * KD + k] : 0.f);
  } else if (i < N_OFF + N_DCN) {
    int j = i - N_OFF;                        // layout [k][n][c], c fastest
    int k = j / (CRP * CRP);
    int r = j % (CRP * CRP);
    int n = r / CRP, c = r % CRP;
    wdc[j] = (__bf16)((n < CRr && c < CRr) ? w_dcn[((size_t)n * CRr + c) * KK + k] : 0.f);
  } else if (i < N_OFF + N_DCN + N_RES) {
    int j = i - N_OFF - N_DCN;                // layout [n][c]
    int n = j / CRP, c = j % CRP;
    wrs[j] = (__bf16)((c < CRr) ? w_res[(size_t)n * CRr + c] : 0.f);
  } else if (i < N_OFF + N_DCN + N_RES + N_RED) {
    int j = i - N_OFF - N_DCN - N_RES;        // layout [n][c], rows 28..31 zero
    int n = j / Cc, c = j % Cc;
    wrd[j] = (__bf16)((n < CRr) ? w_red[(size_t)n * Cc + c] : 0.f);
  }
}

// ---------------- kernel 1: xr = w_red * x  (WMMA, + avg-pool partials) ----
// One wave = 16 consecutive pixels, N = 32 padded channels (two 16-tiles),
// K = 512 in 16 steps of 32. bf16 weights async-staged raw into LDS.
__global__ void reduce_wmma_kernel(const float* __restrict__ x,
                                   float* __restrict__ ws) {
  __shared__ __attribute__((aligned(16))) __bf16 lds_w[CRP * Cc];  // 32 KB, pre-padded

  const __bf16* wrd = (const __bf16*)(ws + WS_WREDB);
#if HAVE_ASYNC_LDS
  for (int i = threadIdx.x; i < CRP * Cc / 8; i += blockDim.x) {
    __builtin_amdgcn_global_load_async_to_lds_b128(
        (__attribute__((address_space(1))) v4i_gcc*)(wrd + i * 8),
        (__attribute__((address_space(3))) v4i_gcc*)(&lds_w[i * 8]),
        0, 0);
  }
#if __has_builtin(__builtin_amdgcn_s_wait_asynccnt)
  __builtin_amdgcn_s_wait_asynccnt(0);
#endif
#else
  for (int i = threadIdx.x; i < CRP * Cc; i += blockDim.x) lds_w[i] = wrd[i];
#endif
  __syncthreads();

  float* xr    = ws + WS_XR;
  float* p_avg = ws + WS_PAVG;

  int wave = blockIdx.x * (blockDim.x >> 5) + (threadIdx.x >> 5);
  int base = wave * 16;                       // global pixel base
  if (base >= Bn * HW) return;
  int ln = lane_id();
  int q = ln & 15;                            // A row / B,D column within tile
  int khf = ln >> 4;                          // lane half -> K grouping
  int b = base / HW, hwb = base % HW;
  const float* xb = x + ((size_t)b * Cc) * HW + hwb;

  v8f acc0 = {}, acc1 = {};
  for (int ks = 0; ks < Cc / 32; ++ks) {
    if (ks + 1 < Cc / 32)
      __builtin_prefetch(xb + (size_t)((ks + 1) * 32) * HW + q, 0, 3);
    v16bf A;
#pragma unroll
    for (int e = 0; e < 16; ++e) {
      int c = ks * 32 + kmap(e, khf);
      A[e] = (__bf16)xb[(size_t)c * HW + q];  // coalesced: 16 consecutive w
    }
    v16bf B0 = load_bfrag(&lds_w[q * Cc + ks * 32], khf);
    v16bf B1 = load_bfrag(&lds_w[(16 + q) * Cc + ks * 32], khf);
    acc0 = wmma_bf16(A, B0, acc0);
    acc1 = wmma_bf16(A, B1, acc1);
  }

  // Padded D columns (28..31) are exactly zero -> store unconditionally.
  float s0 = 0.f, s1 = 0.f;
#pragma unroll
  for (int r = 0; r < 8; ++r) {
    int mm = r + 8 * khf;                     // D row = pixel within tile
    xr[((size_t)(b * CRP + q)) * HW + hwb + mm] = acc0[r];
    xr[((size_t)(b * CRP + 16 + q)) * HW + hwb + mm] = acc1[r];
    s0 += acc0[r];
    s1 += acc1[r];
  }
  atomicAdd(&p_avg[b * CRP + q], s0);
  atomicAdd(&p_avg[b * CRP + 16 + q], s1);
}

// ---------------- kernel 2: per-pixel channel max -> mean over HW ----------
__global__ void pmax_kernel(float* __restrict__ ws) {
  const float* xr = ws + WS_XR;
  float* pmax = ws + WS_PMAX;
  int pix = blockIdx.x * blockDim.x + threadIdx.x;   // 65536
  int b = pix / HW, hw = pix % HW;
  float mx = xr[((size_t)(b * CRP)) * HW + hw];
  for (int c = 1; c < CRr; ++c)
    mx = fmaxf(mx, xr[((size_t)(b * CRP + c)) * HW + hw]);
  __shared__ float red[256];
  red[threadIdx.x] = mx;
  __syncthreads();
  for (int s = 128; s > 0; s >>= 1) {
    if (threadIdx.x < s) red[threadIdx.x] += red[threadIdx.x + s];
    __syncthreads();
  }
  if (threadIdx.x == 0) atomicAdd(&pmax[b], red[0]);   // block spans one b
}

// ---------------- kernel 3: SE gates (tiny MLPs) ---------------------------
__global__ void se_gate_kernel(const float* __restrict__ w1, const float* __restrict__ b1,
                               const float* __restrict__ w2, const float* __restrict__ b2,
                               float* __restrict__ ws) {
  // blockDim.x == Bn*CRr == 448
  const float* p_avg = ws + WS_PAVG;
  const float* pmax  = ws + WS_PMAX;
  float* se = ws + WS_SE;
  __shared__ float h1[Bn * CRr], h2[Bn * CRr];
  int t = threadIdx.x;
  int b = t / CRr, j = t % CRr;
  float accA = b1[j], accM = b1[j];
  float pm = pmax[b] * (1.f / HW);
  for (int i = 0; i < CRr; ++i) {
    float w = w1[j * CRr + i];
    accA += w * (p_avg[b * CRP + i] * (1.f / HW));
    accM += w * pm;
  }
  h1[t] = fmaxf(accA, 0.f);
  h2[t] = fmaxf(accM, 0.f);
  __syncthreads();
  float gA = b2[j], gM = b2[j];
  for (int i = 0; i < CRr; ++i) {
    float w = w2[j * CRr + i];
    gA += w * h1[b * CRr + i];
    gM += w * h2[b * CRr + i];
  }
  se[b * CRP + j] = 1.f / (1.f + __expf(-gA)) + 1.f / (1.f + __expf(-gM));
}

// ---------------- kernel 4: offset conv 7x7 as im2col WMMA GEMM ------------
// M=16 pixels/wave, N=112 padded (7 tiles), K=1376 padded (43 steps of 32).
__global__ void offconv_wmma_kernel(const float* __restrict__ b_off,
                                    float* __restrict__ ws) {
  const float*  xr  = ws + WS_XR;
  const __bf16* wof = (const __bf16*)(ws + WS_WOFFB);
  float* off = ws + WS_OFF;

  int wave = blockIdx.x * (blockDim.x >> 5) + (threadIdx.x >> 5);
  int base = wave * 16;
  if (base >= Bn * HW) return;
  int ln = lane_id();
  int q = ln & 15, khf = ln >> 4;
  int b = base / HW, hwb = base % HW;
  int hh = hwb / Ww, wb = hwb % Ww;

  v8f z = {};
  v8f acc[7];
#pragma unroll
  for (int i = 0; i < 7; ++i) acc[i] = z;

  constexpr int KSTEPS = KDP / 32;  // 43
  for (int ks = 0; ks < KSTEPS; ++ks) {
    v16bf A;
#pragma unroll
    for (int e = 0; e < 16; ++e) {
      int kidx = ks * 32 + kmap(e, khf);
      int kc   = min(kidx, KD - 1);            // clamp for address safety
      int c  = kc / KK;
      int kk = kc - c * KK;
      int r7 = kk / K7, c7 = kk - r7 * K7;
      int py = hh - PADc + r7;
      int px = wb + q - PADc + c7;
      bool valid = (kidx < KD) & (py >= 0) & (py < Hh) & (px >= 0) & (px < Ww);
      int pyc = min(max(py, 0), Hh - 1);
      int pxc = min(max(px, 0), Ww - 1);
      float raw = xr[((size_t)(b * CRP + c)) * HW + pyc * Ww + pxc];
      A[e] = (__bf16)(valid ? raw : 0.f);      // select, not branch
    }
#pragma unroll
    for (int nt = 0; nt < 7; ++nt) {
      int n = nt * 16 + q;
      v16bf Bf = load_bfrag(&wof[(size_t)n * KDP + ks * 32], khf);
      acc[nt] = wmma_bf16(A, Bf, acc[nt]);
    }
  }

#pragma unroll
  for (int nt = 0; nt < 7; ++nt) {
    int n = nt * 16 + q;
    float bo = (n < OFFC) ? b_off[min(n, OFFC - 1)] : 0.f;
#pragma unroll
    for (int r = 0; r < 8; ++r) {
      int mm = r + 8 * khf;
      off[((size_t)(b * OFFP + n)) * HW + hwb + mm] = acc[nt][r] + bo;
    }
  }
}

// ---------------- kernel 5: deformable conv (bilinear gather + WMMA) -------
__global__ void deform_wmma_kernel(float* __restrict__ ws) {
  const float*  xr  = ws + WS_XR;
  const float*  off = ws + WS_OFF;
  const __bf16* wdc = (const __bf16*)(ws + WS_WDCNB);   // [KK][CRP][CRP]
  float* sa    = ws + WS_SA;
  float* bnsum = ws + WS_BNSUM;
  float* bnsq  = ws + WS_BNSQ;

  int wave = blockIdx.x * (blockDim.x >> 5) + (threadIdx.x >> 5);
  int base = wave * 16;
  if (base >= Bn * HW) return;
  int ln = lane_id();
  int q = ln & 15, khf = ln >> 4;
  int b = base / HW, hwb = base % HW;
  int hh = hwb / Ww, wb = hwb % Ww;

  v8f acc0 = {}, acc1 = {};
  for (int k = 0; k < KK; ++k) {
    int r7 = k / K7, c7 = k - r7 * K7;
    float dy = off[((size_t)(b * OFFP + 2 * k)) * HW + hwb + q];
    float dx = off[((size_t)(b * OFFP + 2 * k + 1)) * HW + hwb + q];
    float py = (float)(hh - PADc + r7) + dy;
    float px = (float)(wb + q - PADc + c7) + dx;
    float y0f = floorf(py), x0f = floorf(px);
    float wy = py - y0f, wx = px - x0f;
    int y0 = (int)y0f, x0 = (int)x0f;
    float w00 = (1.f - wy) * (1.f - wx), w01 = (1.f - wy) * wx;
    float w10 = wy * (1.f - wx),        w11 = wy * wx;
    // fold validity into weights, clamp indices so loads stay in-bounds
    int y0c = min(max(y0, 0), Hh - 1),     y1c = min(max(y0 + 1, 0), Hh - 1);
    int x0c = min(max(x0, 0), Ww - 1),     x1c = min(max(x0 + 1, 0), Ww - 1);
    bool vy0 = (y0 >= 0) & (y0 < Hh),      vy1 = (y0 + 1 >= 0) & (y0 + 1 < Hh);
    bool vx0 = (x0 >= 0) & (x0 < Ww),      vx1 = (x0 + 1 >= 0) & (x0 + 1 < Ww);
    if (!(vy0 & vx0)) w00 = 0.f;
    if (!(vy0 & vx1)) w01 = 0.f;
    if (!(vy1 & vx0)) w10 = 0.f;
    if (!(vy1 & vx1)) w11 = 0.f;
    int i00 = y0c * Ww + x0c, i01 = y0c * Ww + x1c;
    int i10 = y1c * Ww + x0c, i11 = y1c * Ww + x1c;

    v16bf A;
#pragma unroll
    for (int e = 0; e < 16; ++e) {
      int c = kmap(e, khf);                 // xr padded: channels 28..31 are 0
      const float* xc = xr + ((size_t)(b * CRP + c)) * HW;
      float v = w00 * xc[i00] + w01 * xc[i01] + w10 * xc[i10] + w11 * xc[i11];
      A[e] = (__bf16)v;
    }
    const __bf16* wk = wdc + (size_t)k * CRP * CRP;
    v16bf B0 = load_bfrag(&wk[q * CRP], khf);
    v16bf B1 = load_bfrag(&wk[(16 + q) * CRP], khf);
    acc0 = wmma_bf16(A, B0, acc0);
    acc1 = wmma_bf16(A, B1, acc1);
  }

  // Padded D columns are exactly zero (zero B columns) -> unconditional.
  float s = 0.f, s2 = 0.f, t = 0.f, t2 = 0.f;
#pragma unroll
  for (int r = 0; r < 8; ++r) {
    int mm = r + 8 * khf;
    float v0 = acc0[r], v1 = acc1[r];
    sa[((size_t)(b * CRP + q)) * HW + hwb + mm] = v0;
    sa[((size_t)(b * CRP + 16 + q)) * HW + hwb + mm] = v1;
    s += v0; s2 += v0 * v0;
    t += v1; t2 += v1 * v1;
  }
  atomicAdd(&bnsum[q], s);
  atomicAdd(&bnsq[q], s2);
  atomicAdd(&bnsum[16 + q], t);
  atomicAdd(&bnsq[16 + q], t2);
}

// ---------------- kernel 6: BN finalize (fold gamma/beta; pads -> 0) -------
__global__ void bn_fin_kernel(const float* __restrict__ gamma,
                              const float* __restrict__ beta,
                              float* __restrict__ ws) {
  int c = threadIdx.x;
  if (c < CRP) {
    if (c < CRr) {
      float N = (float)(Bn * HW);
      float mu  = ws[WS_BNSUM + c] / N;
      float var = ws[WS_BNSQ + c] / N - mu * mu;
      float sc  = gamma[c] * rsqrtf(var + 1e-5f);
      ws[WS_BNSC + c] = sc;
      ws[WS_BNSH + c] = beta[c] - mu * sc;
    } else {
      ws[WS_BNSC + c] = 0.f;
      ws[WS_BNSH + c] = 0.f;
    }
  }
}

// ---------------- kernel 7: blend + expansion GEMM (WMMA) ------------------
// out[b,cout,h,w] = sum_o blend[b,o,h,w] * w_res[cout,o]; N=512 (32 tiles),
// K=28 padded to one 32-step; A fragment (the blend) built once per wave.
__global__ void blend_expand_kernel(const float* __restrict__ wse_p,
                                    const float* __restrict__ wsa_p,
                                    float* __restrict__ out,
                                    float* __restrict__ ws) {
  const float*  xr  = ws + WS_XR;
  const float*  sa  = ws + WS_SA;
  const float*  se  = ws + WS_SE;
  const float*  bsc = ws + WS_BNSC;
  const float*  bsh = ws + WS_BNSH;
  const __bf16* wrs = (const __bf16*)(ws + WS_WRESB);   // [Cc][CRP]

  int wave = blockIdx.x * (blockDim.x >> 5) + (threadIdx.x >> 5);
  int base = wave * 16;
  if (base >= Bn * HW) return;
  int ln = lane_id();
  int q = ln & 15, khf = ln >> 4;
  int b = base / HW, hwb = base % HW;

  float wse = wse_p[0], wsa = wsa_p[0];
  float inv = 1.f / (wse + wsa + 1e-6f);

  // A: padded xr/sa channels are 0 -> blend value is 0 there; unconditional.
  v16bf A;
#pragma unroll
  for (int e = 0; e < 16; ++e) {
    int c = kmap(e, khf);
    size_t idx = ((size_t)(b * CRP + c)) * HW + hwb + q;
    float xrv = xr[idx];
    float bn  = sa[idx] * bsc[c] + bsh[c];
    float rl  = fmaxf(bn, 0.f);
    float sg  = 1.f / (1.f + __expf(-rl));
    float v   = (wse * xrv * se[b * CRP + c] + wsa * xrv * sg) * inv;
    A[e] = (__bf16)v;
  }

  for (int nt = 0; nt < Cc / 16; ++nt) {   // 32 N-tiles
    int n = nt * 16 + q;
    v16bf Bf = load_bfrag(&wrs[(size_t)n * CRP], khf);
    v8f d = {};
    d = wmma_bf16(A, Bf, d);
#pragma unroll
    for (int r = 0; r < 8; ++r) {
      int mm = r + 8 * khf;
      out[((size_t)(b * Cc + n)) * HW + hwb + mm] = d[r];
    }
  }
}

// ---------------- launcher ----------------
extern "C" void kernel_launch(void* const* d_in, const int* in_sizes, int n_in,
                              void* d_out, int out_size, void* d_ws, size_t ws_size,
                              hipStream_t stream) {
  const float* x        = (const float*)d_in[0];
  const float* w_red    = (const float*)d_in[1];
  const float* w_se1    = (const float*)d_in[2];
  const float* b_se1    = (const float*)d_in[3];
  const float* w_se2    = (const float*)d_in[4];
  const float* b_se2    = (const float*)d_in[5];
  const float* w_off    = (const float*)d_in[6];
  const float* b_off    = (const float*)d_in[7];
  const float* w_dcn    = (const float*)d_in[8];
  const float* bn_gamma = (const float*)d_in[9];
  const float* bn_beta  = (const float*)d_in[10];
  const float* w_res    = (const float*)d_in[11];
  const float* w_se_s   = (const float*)d_in[12];
  const float* w_sa_s   = (const float*)d_in[13];
  float* out = (float*)d_out;
  float* ws  = (float*)d_ws;

  constexpr int TILES = Bn * HW / 16;       // 4096 pixel tiles
  constexpr int WPB   = 8;                  // waves per 256-thread block
  dim3 blk(256);
  dim3 gemm_grid(TILES / WPB);              // 512 blocks

  constexpr int PADN = OFFP * KDP + KK * CRP * CRP + Cc * CRP + CRP * Cc; // 237,056

  zero_acc_kernel<<<2, blk, 0, stream>>>(ws);
  pad_weights_kernel<<<(PADN + 255) / 256, blk, 0, stream>>>(w_off, w_dcn, w_res, w_red, ws);
  reduce_wmma_kernel<<<gemm_grid, blk, 0, stream>>>(x, ws);
  pmax_kernel<<<(Bn * HW) / 256, blk, 0, stream>>>(ws);
  se_gate_kernel<<<1, Bn * CRr, 0, stream>>>(w_se1, b_se1, w_se2, b_se2, ws);
  offconv_wmma_kernel<<<gemm_grid, blk, 0, stream>>>(b_off, ws);
  deform_wmma_kernel<<<gemm_grid, blk, 0, stream>>>(ws);
  bn_fin_kernel<<<1, 32, 0, stream>>>(bn_gamma, bn_beta, ws);
  blend_expand_kernel<<<gemm_grid, blk, 0, stream>>>(w_se_s, w_sa_s, out, ws);
}
